// OuterProduct_80092550136283
// MI455X (gfx1250) — matrix operand order repaired and checked by reading, not compiled, not run on hardware
//
#include <hip/hip_runtime.h>

typedef __attribute__((ext_vector_type(16))) _Float16 v16h;
typedef __attribute__((ext_vector_type(8)))  _Float16 v8h;
typedef __attribute__((ext_vector_type(8)))  float    v8f;

constexpr int Nn = 384;   // sequence length
constexpr int Dd = 384;   // input feature dim
constexpr int Cc = 32;    // projection channels (== WMMA K!)
constexpr int Oo = 128;   // output channels

// ---------------------------------------------------------------------------
// Kernel 1: left = mask*(seq_act@left_w + left_b)  (kept f32, feeds VALU GEMV)
//           right = mask*(seq_act@right_w + right_b) (stored f16, feeds WMMA A)
// One thread per (side, row, channel): 2*384*32 = 24576 threads.
// ---------------------------------------------------------------------------
__global__ void __launch_bounds__(256)
proj_kernel(const float* __restrict__ seq_act,
            const float* __restrict__ seq_mask,
            const float* __restrict__ left_w,
            const float* __restrict__ left_b,
            const float* __restrict__ right_w,
            const float* __restrict__ right_b,
            float*    __restrict__ leftF,
            _Float16* __restrict__ rightH) {
  const int g    = blockIdx.x * blockDim.x + threadIdx.x;   // [0, 24576)
  const int side = g / (Nn * Cc);
  const int rc   = g % (Nn * Cc);
  const int r    = rc / Cc;
  const int c    = rc % Cc;

  const float* w  = side ? right_w : left_w;
  const float* bb = side ? right_b : left_b;
  const float* arow = seq_act + (size_t)r * Dd;

  float acc = 0.0f;
#pragma unroll 4
  for (int d = 0; d < Dd; ++d)
    acc = fmaf(arow[d], w[d * Cc + c], acc);

  const float val = seq_mask[r] * (acc + bb[c]);
  if (side) rightH[rc] = (_Float16)val;
  else      leftF[rc]  = val;
}

// ---------------------------------------------------------------------------
// Kernel 2: one block per b (384 blocks, 256 threads = 8 waves).
//   Phase 1 (VALU GEMV): tmpS[e*128+o] = sum_c left[b,c]*out_w[c*4096+e*128+o]
//   Phase 2 (WMMA): wave w owns o-tile w; B operand (32x16 of tmpS) kept in
//   registers across all 24 d-tiles; 1 WMMA per 16x16 output tile (K=32=C).
// ---------------------------------------------------------------------------
__global__ void __launch_bounds__(256)
outer_kernel(const float* __restrict__ out_w,
             const float* __restrict__ out_b,
             const float* __restrict__ leftF,
             const _Float16* __restrict__ rightH,
             float* __restrict__ out) {
  __shared__ float    leftS[Cc];
  __shared__ _Float16 tmpS[Cc * Oo];          // 32x128 f16 = 8 KB

  const int b   = blockIdx.x;
  const int tid = threadIdx.x;

  if (tid < Cc) leftS[tid] = leftF[b * Cc + tid];
  __syncthreads();

  // ---- Phase 1: GEMV, 16 contiguous outputs per thread (coalesced W reads) --
  {
    const int n0 = tid * 16;                  // 256 threads * 16 = 4096
    float acc[16];
#pragma unroll
    for (int i = 0; i < 16; ++i) acc[i] = 0.0f;
#pragma unroll 4
    for (int c = 0; c < Cc; ++c) {
      const float lw = leftS[c];
      const float* Wr = out_w + (size_t)c * (Cc * Oo) + n0;
#pragma unroll
      for (int i = 0; i < 16; ++i) acc[i] = fmaf(lw, Wr[i], acc[i]);
    }
#pragma unroll
    for (int i = 0; i < 16; ++i) tmpS[n0 + i] = (_Float16)acc[i];
  }
  __syncthreads();

  // ---- Phase 2: out[b, d, o] = right(384x32) @ tmpS(32x128) + out_b --------
  const int wave   = tid >> 5;                // o-tile index, 0..7
  const int lane   = tid & 31;
  const int hi     = lane >> 4;               // 0 = lanes 0-15, 1 = lanes 16-31
  const int cIdx   = lane & 15;
  const int col    = wave * 16 + cIdx;        // output channel o

  // B operand: this code's contiguous k order — lanes 0-15 hold K=0..15 of column N=lane,
  // lanes 16-31 hold K=16..31 of column N=lane-16.
  const int kb = hi * 16;
  v16h bv;
#pragma unroll
  for (int j = 0; j < 16; ++j)
    bv[j] = tmpS[(kb + j) * Oo + col];

  const float bias = out_b[col];

  // A operand: this code's contiguous k order — lane<16: elems 0..15 = K0..15;
  // lane>=16: elems 0..15 = K16..31 (matches B).  Row = tile*16 + (lane&15).
  const int k0 = hi * 16;

#pragma unroll 2
  for (int dT = 0; dT < Nn / 16; ++dT) {
    const int row = dT * 16 + cIdx;
    const _Float16* ap = rightH + row * Cc;
    const v8h a0 = *(const v8h*)(ap + k0);        // 16B-aligned
    const v8h a1 = *(const v8h*)(ap + k0 + 8);    // 16B-aligned
    v16h av;
#pragma unroll
    for (int j = 0; j < 8; ++j) { av[j] = a0[j]; av[8 + j] = a1[j]; }

    v8f acc = {};
    acc = __builtin_amdgcn_wmma_f32_16x16x32_f16(
        /*neg_a=*/false, av, /*neg_b=*/false, bv,
        /*c_mod=*/(short)0, acc, /*reuse_a=*/false, /*reuse_b=*/false);

    // C/D layout: element v -> row v + 8*hi, col = lane&15.
    float* orow = out + (((size_t)b * Nn) + dT * 16 + hi * 8) * Oo + col;
#pragma unroll
    for (int v = 0; v < 8; ++v)
      orow[(size_t)v * Oo] = acc[v] + bias;
  }
}

// ---------------------------------------------------------------------------
extern "C" void kernel_launch(void* const* d_in, const int* in_sizes, int n_in,
                              void* d_out, int out_size, void* d_ws, size_t ws_size,
                              hipStream_t stream) {
  const float* seq_act  = (const float*)d_in[0];
  const float* seq_mask = (const float*)d_in[1];
  const float* left_w   = (const float*)d_in[2];
  const float* left_b   = (const float*)d_in[3];
  const float* right_w  = (const float*)d_in[4];
  const float* right_b  = (const float*)d_in[5];
  const float* out_w    = (const float*)d_in[6];
  const float* out_b    = (const float*)d_in[7];
  float* out = (float*)d_out;

  // Workspace: left (f32) then right (f16)
  float*    leftF  = (float*)d_ws;
  _Float16* rightH = (_Float16*)((char*)d_ws + (size_t)Nn * Cc * sizeof(float));

  proj_kernel<<<(2 * Nn * Cc) / 256, 256, 0, stream>>>(
      seq_act, seq_mask, left_w, left_b, right_w, right_b, leftF, rightH);

  outer_kernel<<<Nn, 256, 0, stream>>>(out_w, out_b, leftF, rightH, out);
}
